// VectorQuantizer_33139967656240
// MI455X (gfx1250) — matrix operand order repaired and checked by reading, not compiled, not run on hardware
//
#include <hip/hip_runtime.h>

typedef __attribute__((ext_vector_type(16))) __bf16 v16bf;
typedef __attribute__((ext_vector_type(8)))  float  v8f;
typedef int v4i_ls __attribute__((vector_size(16)));   // builtin's pointee type

#define KE        1024      // codebook entries
#define DIMS      128       // embedding dim
#define NROWS     32768     // flattened rows of x
#define M_TILE    128       // rows per block (8 waves x 16 rows)
#define NCB_TILE  16        // codebook entries per pipeline chunk
#define NCHUNK    (KE / NCB_TILE)
#define QN        (NROWS * DIMS)   // 4194304 floats of quantized output

#if defined(__has_builtin)
#if __has_builtin(__builtin_amdgcn_global_load_async_to_lds_b128)
#define ASYNC_LDS 1
#endif
#endif

// ---------------------------------------------------------------------------
// Stage one 16-entry codebook chunk (hi 4KB + lo 4KB) into LDS.
// 256 threads x 16B each per matrix -> 2 async instructions per wave per chunk.
// ---------------------------------------------------------------------------
__device__ __forceinline__ void stage_chunk(const __bf16* __restrict__ gh,
                                            const __bf16* __restrict__ gl,
                                            __bf16* lh, __bf16* ll, int tid) {
#if ASYNC_LDS
  __builtin_amdgcn_global_load_async_to_lds_b128((v4i_ls*)(gh + tid * 8),
                                                 (v4i_ls*)(lh + tid * 8), 0, 0);
  __builtin_amdgcn_global_load_async_to_lds_b128((v4i_ls*)(gl + tid * 8),
                                                 (v4i_ls*)(ll + tid * 8), 0, 0);
#else
  ((float4*)lh)[tid] = ((const float4*)gh)[tid];
  ((float4*)ll)[tid] = ((const float4*)gl)[tid];
#endif
}

__device__ __forceinline__ void wait_stage() {
#if ASYNC_LDS
  __builtin_amdgcn_s_wait_asynccnt(0);
#endif
}

// ---------------------------------------------------------------------------
// Kernel 1: emb (f32) -> bf16 hi/lo split + exact f32 row sums of squares.
// One wave per codebook entry: 128 blocks x 8 waves.
// ---------------------------------------------------------------------------
__global__ void __launch_bounds__(256) vq_prep(const float* __restrict__ emb,
                                               float* __restrict__ esq,
                                               __bf16* __restrict__ ehi,
                                               __bf16* __restrict__ elo) {
  const int lane = threadIdx.x & 31;
  const int wave = threadIdx.x >> 5;
  const int e = blockIdx.x * 8 + wave;           // 0..1023
  const int base = e * DIMS + lane * 4;
  const float4 v = *(const float4*)(emb + base);
  float f[4] = {v.x, v.y, v.z, v.w};
  float s = 0.f;
#pragma unroll
  for (int j = 0; j < 4; ++j) {
    s = fmaf(f[j], f[j], s);
    __bf16 h = (__bf16)f[j];
    ehi[base + j] = h;
    elo[base + j] = (__bf16)(f[j] - (float)h);
  }
#pragma unroll
  for (int d = 1; d < 32; d <<= 1) s += __shfl_xor(s, d, 32);
  if (lane == 0) esq[e] = s;
}

// ---------------------------------------------------------------------------
// Kernel 2: fused distance-GEMM + argmin + gather + partial MSE.
// Grid: 256 blocks x 256 threads. Wave w owns rows [128*b + 16*w, +16).
// ---------------------------------------------------------------------------
__global__ void __launch_bounds__(256) vq_main(const float* __restrict__ x,
                                               const float* __restrict__ emb,
                                               const float* __restrict__ esq,
                                               const __bf16* __restrict__ ehi,
                                               const __bf16* __restrict__ elo,
                                               float* __restrict__ out,
                                               float* __restrict__ partials) {
  __shared__ __align__(16) __bf16 sBH[2][NCB_TILE * DIMS];   // 2 x 4KB
  __shared__ __align__(16) __bf16 sBL[2][NCB_TILE * DIMS];   // 2 x 4KB
  __shared__ float sEsq[KE];                                  // 4KB
  __shared__ int   sIdx[M_TILE];
  __shared__ float sRed[8];

  const int tid  = threadIdx.x;
  const int lane = tid & 31;
  const int wave = tid >> 5;
  const int sel  = lane >> 4;      // 0: lanes 0-15, 1: lanes 16-31
  const int l16  = lane & 15;

  // e_sq -> LDS (visible after first pipeline barrier)
#pragma unroll
  for (int i = 0; i < 4; ++i) sEsq[tid + 256 * i] = esq[tid + 256 * i];

  // -------- A fragments: this lane's row, bf16 hi/lo split, in registers ----
  // A 16x32 bf16, this code's k order (as B): lane M = lane%16; lanes 0-15 carry K = kk+0..15,
  // lanes 16-31 carry K = kk+16..31.
  const int mrow = blockIdx.x * M_TILE + wave * 16 + l16;
  const float4* xr = (const float4*)(x + (size_t)mrow * DIMS);
  v16bf a_hi[4], a_lo[4];
#pragma unroll
  for (int k4 = 0; k4 < 4; ++k4) {
    const int q0 = k4 * 8 + sel * 4;             // float4 index of first chunk
    float4 f0 = xr[q0], f1 = xr[q0 + 1], f2 = xr[q0 + 2], f3 = xr[q0 + 3];
    float fs[16] = {f0.x, f0.y, f0.z, f0.w, f1.x, f1.y, f1.z, f1.w,
                    f2.x, f2.y, f2.z, f2.w, f3.x, f3.y, f3.z, f3.w};
#pragma unroll
    for (int j = 0; j < 16; ++j) {
      __bf16 h = (__bf16)fs[j];
      a_hi[k4][j] = h;
      a_lo[k4][j] = (__bf16)(fs[j] - (float)h);
    }
  }

  float best[8];
  int   bestn[8];
#pragma unroll
  for (int i = 0; i < 8; ++i) { best[i] = 3.4e38f; bestn[i] = 0; }

  // -------- pipelined codebook sweep ---------------------------------------
  stage_chunk(ehi, elo, sBH[0], sBL[0], tid);
  for (int c = 0; c < NCHUNK; ++c) {
    wait_stage();          // this wave's chunk-c async loads landed in LDS
    __syncthreads();       // everyone's chunk c visible; all done with c-1
    if (c + 1 < NCHUNK)
      stage_chunk(ehi + (c + 1) * NCB_TILE * DIMS,
                  elo + (c + 1) * NCB_TILE * DIMS,
                  sBH[(c + 1) & 1], sBL[(c + 1) & 1], tid);

    const __bf16* bh_base = sBH[c & 1];
    const __bf16* bl_base = sBL[c & 1];
    // B 32x16 bf16, this code's k order (as A): lane N = lane%16 (codebook entry), lanes 0-15 hold
    // K=kk+0..15, lanes 16-31 hold K=kk+16..31 -> contiguous 32B of emb row.
    const int eoff = l16 * DIMS + sel * 16;
    v8f acc = {};
#pragma unroll
    for (int k4 = 0; k4 < 4; ++k4) {
      v16bf bh = *(const v16bf*)(bh_base + eoff + k4 * 32);
      v16bf bl = *(const v16bf*)(bl_base + eoff + k4 * 32);
      acc = __builtin_amdgcn_wmma_f32_16x16x32_bf16(false, a_hi[k4], false, bh,
                                                    (short)0, acc, false, false);
      acc = __builtin_amdgcn_wmma_f32_16x16x32_bf16(false, a_lo[k4], false, bh,
                                                    (short)0, acc, false, false);
      acc = __builtin_amdgcn_wmma_f32_16x16x32_bf16(false, a_hi[k4], false, bl,
                                                    (short)0, acc, false, false);
    }
    const int eg = c * NCB_TILE + l16;           // codebook index of this lane
    const float es = sEsq[eg];
#pragma unroll
    for (int i = 0; i < 8; ++i) {
      float s = fmaf(-2.0f, acc[i], es);         // ||e||^2 - 2 x.e
      if (s < best[i]) { best[i] = s; bestn[i] = eg; }
    }
  }

  // -------- argmin reduction across the 16 N-lanes of each half-wave -------
#pragma unroll
  for (int d = 1; d < 16; d <<= 1) {
#pragma unroll
    for (int i = 0; i < 8; ++i) {
      float ob = __shfl_xor(best[i], d, 32);
      int   on = __shfl_xor(bestn[i], d, 32);
      if (ob < best[i] || (ob == best[i] && on < bestn[i])) {
        best[i] = ob; bestn[i] = on;
      }
    }
  }
  // C layout: VGPR i holds M = i (lanes 0-15) or M = 8+i (lanes 16-31)
  if (l16 == 0) {
#pragma unroll
    for (int i = 0; i < 8; ++i) sIdx[wave * 16 + sel * 8 + i] = bestn[i];
  }
  __syncthreads();

  // -------- phase 2: exact f32 gather, quantized output, MSE partial -------
  const int r  = tid >> 1;                       // row within block, 0..127
  const int hc = (tid & 1) * 64;                 // column half
  const int grow = blockIdx.x * M_TILE + r;
  const int eidx = sIdx[r];
  const float4* q4  = (const float4*)(emb + (size_t)eidx * DIMS + hc);
  const float4* x4b = (const float4*)(x + (size_t)grow * DIMS + hc);
  float4* o4 = (float4*)(out + (size_t)grow * DIMS + hc);
  float ss = 0.f;
#pragma unroll
  for (int j = 0; j < 16; ++j) {
    float4 q = q4[j], xv = x4b[j];
    float dx = q.x - xv.x, dy = q.y - xv.y, dz = q.z - xv.z, dw = q.w - xv.w;
    ss += dx * dx + dy * dy + dz * dz + dw * dw;
    o4[j] = q;
  }
  // encoding indices (as float, concatenated after quantized + loss)
  if (tid < M_TILE)
    out[QN + 1 + blockIdx.x * M_TILE + tid] = (float)sIdx[tid];

#pragma unroll
  for (int d = 1; d < 32; d <<= 1) ss += __shfl_xor(ss, d, 32);
  if (lane == 0) sRed[wave] = ss;
  __syncthreads();
  if (tid == 0) {
    float t = 0.f;
#pragma unroll
    for (int i = 0; i < 8; ++i) t += sRed[i];
    partials[blockIdx.x] = t;
  }
}

// ---------------------------------------------------------------------------
// Kernel 3: deterministic reduction of 256 block partials -> loss scalar.
// loss = (1 + 0.25) * mean((q - x)^2)
// ---------------------------------------------------------------------------
__global__ void __launch_bounds__(256) vq_loss(const float* __restrict__ partials,
                                               float* __restrict__ out_loss) {
  __shared__ float red[8];
  const int lane = threadIdx.x & 31;
  const int wave = threadIdx.x >> 5;
  float s = partials[threadIdx.x];
#pragma unroll
  for (int d = 1; d < 32; d <<= 1) s += __shfl_xor(s, d, 32);
  if (lane == 0) red[wave] = s;
  __syncthreads();
  if (threadIdx.x == 0) {
    float t = 0.f;
#pragma unroll
    for (int i = 0; i < 8; ++i) t += red[i];
    *out_loss = 1.25f * t / 4194304.0f;
  }
}

// ---------------------------------------------------------------------------
extern "C" void kernel_launch(void* const* d_in, const int* in_sizes, int n_in,
                              void* d_out, int out_size, void* d_ws, size_t ws_size,
                              hipStream_t stream) {
  const float* x   = (const float*)d_in[0];   // [32,32,32,128] f32
  const float* emb = (const float*)d_in[1];   // [1024,128] f32
  float* out = (float*)d_out;                 // [QN quantized | loss | 32768 idx]
  char* ws = (char*)d_ws;
  // ws layout: e_sq 4KB | emb_hi 256KB | emb_lo 256KB | partials 1KB (~517KB)
  float*  esq      = (float*)ws;
  __bf16* ehi      = (__bf16*)(ws + 4096);
  __bf16* elo      = (__bf16*)(ws + 4096 + 262144);
  float*  partials = (float*)(ws + 4096 + 2 * 262144);

  vq_prep<<<128, 256, 0, stream>>>(emb, esq, ehi, elo);
  vq_main<<<256, 256, 0, stream>>>(x, emb, esq, ehi, elo, out, partials);
  vq_loss<<<1, 256, 0, stream>>>(partials, out + QN);
}
